// SelfAttention_21655225106846
// MI455X (gfx1250) — matrix operand order repaired and checked by reading, not compiled, not run on hardware
//
#include <hip/hip_runtime.h>
#include <hip/hip_bf16.h>

// ---------------------------------------------------------------------------
// Types
// ---------------------------------------------------------------------------
typedef __bf16 bf16;
typedef __attribute__((ext_vector_type(4)))  float v4f;
typedef __attribute__((ext_vector_type(8)))  float v8f;
typedef __attribute__((ext_vector_type(8)))  bf16  v8bf;
typedef __attribute__((ext_vector_type(16))) bf16  v16bf;

#define B_SZ   4
#define L_SZ   4096
#define C_SZ   512
#define DK_SZ  64
#define OUT_SZ 512

// ---------------------------------------------------------------------------
// Helpers
// ---------------------------------------------------------------------------
__device__ __forceinline__ v16bf join16(v8bf lo, v8bf hi) {
  return __builtin_shufflevector(lo, hi, 0, 1, 2, 3, 4, 5, 6, 7,
                                          8, 9, 10, 11, 12, 13, 14, 15);
}

__device__ __forceinline__ v8bf load8bf(const bf16* p) {
  return *(const v8bf*)p;                       // 16B -> global_load_b128
}

__device__ __forceinline__ v8f wmma_bf16(v16bf a, v16bf b, v8f c) {
  return __builtin_amdgcn_wmma_f32_16x16x32_bf16(
      /*neg_a=*/false, a, /*neg_b=*/false, b,
      /*c_mod=*/(short)0, c, /*reuse_a=*/false, /*reuse_b=*/false);
}

// ---------------------------------------------------------------------------
// Kernel 0: one-time f32 -> bf16 conversion (bandwidth bound, b128 in/out).
// Removes all converts from the GEMM hot loops.
// ---------------------------------------------------------------------------
__global__ void __launch_bounds__(256)
cvt_f32_to_bf16_kernel(const float* __restrict__ src, bf16* __restrict__ dst,
                       int n8) {
  const int i = blockIdx.x * blockDim.x + threadIdx.x;
  if (i >= n8) return;
  const float* p = src + (size_t)i * 8;
  v4f a = *(const v4f*)p;
  v4f b = *(const v4f*)(p + 4);
  v8bf r;
  r[0] = (bf16)a[0]; r[1] = (bf16)a[1]; r[2] = (bf16)a[2]; r[3] = (bf16)a[3];
  r[4] = (bf16)b[0]; r[5] = (bf16)b[1]; r[6] = (bf16)b[2]; r[7] = (bf16)b[3];
  *(v8bf*)(dst + (size_t)i * 8) = r;
}

// ---------------------------------------------------------------------------
// Kernel 1: QKV projection, pure bf16 WMMA.  Y = x @ W^T + b.
// wb holds [Wq(64) | Wk(64) | Wv(512)] rows, bf16, row-major over C.
// Each wave computes a 16(M) x 64(N) tile; inner loop per 32-wide k-step is
// 10x global_load_b128 + 4x v_wmma (no converts).
// Q,K stored bf16 [B*L, 64]; V stored bf16 transposed Vt[b][o][l].
// ---------------------------------------------------------------------------
__global__ void __launch_bounds__(128)
qkv_proj_kernel(const bf16* __restrict__ xb,
                const bf16* __restrict__ wb,
                const float* __restrict__ bq,
                const float* __restrict__ bk,
                const float* __restrict__ bv,
                bf16* __restrict__ qb,
                bf16* __restrict__ kb,
                bf16* __restrict__ vt) {
  const int lane = threadIdx.x & 31;
  const int wave = threadIdx.x >> 5;
  const int l16  = lane & 15;
  const int half = lane >> 4;          // 0: lanes 0-15, 1: lanes 16-31

  const int m0 = blockIdx.x * 64 + wave * 16;   // rows of x (B*L flattened)
  const int n0 = blockIdx.y * 64;               // global output column tile

  v8f acc[4] = {};

  const bf16* xrow = xb + (size_t)(m0 + l16) * C_SZ;
  const int aoff = half ? 16 : 0;      // A-layout: lanes>=16 hold K=16..31
  const int boff = half ? 16 : 0;      // B-layout: lanes>=16 hold K=16..31

  for (int c = 0; c < C_SZ; c += 32) {
    // A fragment: x[m, c..c+31] (16x32 bf16)
    v16bf a = join16(load8bf(xrow + c + aoff), load8bf(xrow + c + aoff + 8));
#pragma unroll
    for (int t = 0; t < 4; ++t) {
      // B fragment: Bt[n][k] = W[n, c+k] -> row-major W rows, contiguous
      const bf16* wrow = wb + (size_t)(n0 + t * 16 + l16) * C_SZ + c + boff;
      v16bf bfrag = join16(load8bf(wrow), load8bf(wrow + 8));
      acc[t] = wmma_bf16(a, bfrag, acc[t]);
    }
  }

  // bias select for this 64-aligned N tile
  const float* bias;  int nl0;
  if (n0 < 64)        { bias = bq; nl0 = n0; }
  else if (n0 < 128)  { bias = bk; nl0 = n0 - 64; }
  else                { bias = bv; nl0 = n0 - 128; }

  // write out.  D layout: VGPR r holds row r (lanes 0-15) or 8+r (lanes
  // 16-31); column = lane%16 (+16 per tile step).
  const int hi8 = half * 8;
#pragma unroll
  for (int t = 0; t < 4; ++t) {
    const int ncol = nl0 + t * 16 + l16;
    const float bval = bias[ncol];
    if (n0 < 128) {
      bf16* dst = (n0 < 64) ? qb : kb;
#pragma unroll
      for (int r = 0; r < 8; ++r) {
        const int m = m0 + r + hi8;
        dst[(size_t)m * DK_SZ + ncol] = (bf16)(acc[t][r] + bval);
      }
    } else {
#pragma unroll
      for (int r = 0; r < 8; ++r) {
        const int m    = m0 + r + hi8;
        const int bidx = m >> 12;        // / L_SZ
        const int lidx = m & (L_SZ - 1); // % L_SZ
        vt[((size_t)bidx * OUT_SZ + ncol) * L_SZ + lidx] =
            (bf16)(acc[t][r] + bval);
      }
    }
  }
}

// ---------------------------------------------------------------------------
// Kernel 2: causal flash attention.
// Block = 4 waves.  All waves share the same 16 query rows (q0..q0+15);
// wave w owns output columns [w*128, w*128+128).  Online softmax, key
// blocks of 32, S and P*V both on v_wmma_f32_16x16x32_bf16.
// ---------------------------------------------------------------------------
__global__ void __launch_bounds__(128)
flash_attn_kernel(const bf16* __restrict__ qb,
                  const bf16* __restrict__ kb,
                  const bf16* __restrict__ vt,
                  float* __restrict__ out) {
  __shared__ bf16 lds_p[4][16 * 32];   // per-wave P tile (16 q x 32 k)

  const int lane = threadIdx.x & 31;
  const int wave = threadIdx.x >> 5;
  const int l16  = lane & 15;
  const int half = lane >> 4;
  const int hi8  = half * 8;

  const int q0 = blockIdx.x * 16;
  const int b  = blockIdx.y;
  const int o0 = wave * 128;

  const bf16* Q = qb + (size_t)b * L_SZ * DK_SZ;
  const bf16* K = kb + (size_t)b * L_SZ * DK_SZ;
  const bf16* V = vt + (size_t)b * OUT_SZ * L_SZ;   // transposed [o][l]

  // Q A-fragments for dk chunks [0,32) and [32,64), kept for whole loop
  const bf16* qrow = Q + (size_t)(q0 + l16) * DK_SZ;
  const int aoff = half ? 16 : 0;
  const v16bf qa0 = join16(load8bf(qrow + aoff),      load8bf(qrow + aoff + 8));
  const v16bf qa1 = join16(load8bf(qrow + 32 + aoff), load8bf(qrow + 32 + aoff + 8));

  float mstate[8], lstate[8];
#pragma unroll
  for (int r = 0; r < 8; ++r) { mstate[r] = -3.0e38f; lstate[r] = 0.0f; }
  v8f oacc[8] = {};

  const int koff = half ? 16 : 0;
  const int jend = q0 + 16;            // causal: need keys <= q0+15

  for (int j0 = 0; j0 < jend; j0 += 32) {
    if (j0 + 32 < jend)                // pull next K block toward the WGP
      __builtin_prefetch(K + (size_t)(j0 + 32 + l16) * DK_SZ, 0, 3);

    // ---- S = Q @ K^T for keys [j0, j0+32) : two 16x16 tiles --------------
    v8f s0 = {}, s1 = {};
    {
      const bf16* kr0 = K + (size_t)(j0 + l16) * DK_SZ + koff;        // keys j0..+15
      const bf16* kr1 = K + (size_t)(j0 + 16 + l16) * DK_SZ + koff;   // keys +16..+31
      v16bf f = join16(load8bf(kr0), load8bf(kr0 + 8));          // dk 0..31
      s0 = wmma_bf16(qa0, f, s0);
      f = join16(load8bf(kr0 + 32), load8bf(kr0 + 40));          // dk 32..63
      s0 = wmma_bf16(qa1, f, s0);
      f = join16(load8bf(kr1), load8bf(kr1 + 8));
      s1 = wmma_bf16(qa0, f, s1);
      f = join16(load8bf(kr1 + 32), load8bf(kr1 + 40));
      s1 = wmma_bf16(qa1, f, s1);
    }

    // ---- scale + causal mask + online softmax ----------------------------
    const float scale = 0.125f;        // 1/sqrt(64)
    float p0[8], p1[8], alpha[8];
#pragma unroll
    for (int r = 0; r < 8; ++r) {
      const int qg  = q0 + r + hi8;    // this element's query row
      const int k0i = j0 + l16;        // key col, subtile 0
      const int k1i = k0i + 16;        // key col, subtile 1
      float v0 = (k0i <= qg) ? s0[r] * scale : -3.0e38f;
      float v1 = (k1i <= qg) ? s1[r] * scale : -3.0e38f;
      // row-max across the 16 lanes of this half-wave
      float mx = fmaxf(v0, v1);
      mx = fmaxf(mx, __shfl_xor(mx, 1, 32));
      mx = fmaxf(mx, __shfl_xor(mx, 2, 32));
      mx = fmaxf(mx, __shfl_xor(mx, 4, 32));
      mx = fmaxf(mx, __shfl_xor(mx, 8, 32));
      const float mnew = fmaxf(mstate[r], mx);
      const float a = __expf(mstate[r] - mnew);
      const float e0 = __expf(v0 - mnew);
      const float e1 = __expf(v1 - mnew);
      float rs = e0 + e1;
      rs += __shfl_xor(rs, 1, 32);
      rs += __shfl_xor(rs, 2, 32);
      rs += __shfl_xor(rs, 4, 32);
      rs += __shfl_xor(rs, 8, 32);
      lstate[r] = lstate[r] * a + rs;
      mstate[r] = mnew;
      alpha[r]  = a;
      p0[r] = e0;
      p1[r] = e1;
    }

    // rescale running output accumulators (row = r + hi8, same per element)
#pragma unroll
    for (int t = 0; t < 8; ++t)
#pragma unroll
      for (int r = 0; r < 8; ++r)
        oacc[t][r] *= alpha[r];

    // ---- P (C/D layout) -> LDS -> A layout -------------------------------
    bf16* pb = lds_p[wave];
#pragma unroll
    for (int r = 0; r < 8; ++r) {
      const int ml = r + hi8;
      pb[ml * 32 + l16]      = (bf16)p0[r];
      pb[ml * 32 + 16 + l16] = (bf16)p1[r];
    }
    __builtin_amdgcn_wave_barrier();   // DS ops are in-order per wave
    const bf16* prow = pb + l16 * 32 + aoff;
    const v16bf pa = join16(load8bf(prow), load8bf(prow + 8));

    // ---- O += P @ V : 8 n-tiles of 16 output columns ---------------------
#pragma unroll
    for (int t = 0; t < 8; ++t) {
      const bf16* vr = V + (size_t)(o0 + t * 16 + l16) * L_SZ + j0 + koff;
      v16bf vf = join16(load8bf(vr), load8bf(vr + 8));
      oacc[t] = wmma_bf16(pa, vf, oacc[t]);
    }
  }

  // ---- normalize and store ----------------------------------------------
#pragma unroll
  for (int r = 0; r < 8; ++r) {
    const float inv = 1.0f / lstate[r];
    const int m = q0 + r + hi8;
    float* orow = out + ((size_t)b * L_SZ + m) * OUT_SZ + o0;
#pragma unroll
    for (int t = 0; t < 8; ++t)
      orow[t * 16 + l16] = oacc[t][r] * inv;
  }
}

// ---------------------------------------------------------------------------
// Launch
// ---------------------------------------------------------------------------
extern "C" void kernel_launch(void* const* d_in, const int* in_sizes, int n_in,
                              void* d_out, int out_size, void* d_ws,
                              size_t ws_size, hipStream_t stream) {
  const float* x  = (const float*)d_in[0];
  const float* Wq = (const float*)d_in[1];
  const float* bq = (const float*)d_in[2];
  const float* Wk = (const float*)d_in[3];
  const float* bk = (const float*)d_in[4];
  const float* Wv = (const float*)d_in[5];
  const float* bv = (const float*)d_in[6];
  float* out = (float*)d_out;

  // workspace carve (bf16):
  //   xb [B*L, C]      16.8 MB
  //   wb [640, C]       0.65 MB  (Wq rows 0-63 | Wk 64-127 | Wv 128-639)
  //   qb [B*L, 64]      2 MB
  //   kb [B*L, 64]      2 MB
  //   vt [B, 512, L]   16.8 MB   (V transposed)
  const size_t x_elems  = (size_t)B_SZ * L_SZ * C_SZ;
  const size_t w_elems  = (size_t)(2 * DK_SZ + OUT_SZ) * C_SZ;
  const size_t qk_elems = (size_t)B_SZ * L_SZ * DK_SZ;
  bf16* xb = (bf16*)d_ws;
  bf16* wb = xb + x_elems;
  bf16* qb = wb + w_elems;
  bf16* kb = qb + qk_elems;
  bf16* vt = kb + qk_elems;

  // one-time conversions (bandwidth bound)
  {
    const int n8x = (int)(x_elems / 8);
    cvt_f32_to_bf16_kernel<<<(n8x + 255) / 256, 256, 0, stream>>>(x, xb, n8x);
    const int n8qk = (DK_SZ * C_SZ) / 8;           // 4096
    cvt_f32_to_bf16_kernel<<<(n8qk + 255) / 256, 256, 0, stream>>>(
        Wq, wb, n8qk);
    cvt_f32_to_bf16_kernel<<<(n8qk + 255) / 256, 256, 0, stream>>>(
        Wk, wb + (size_t)DK_SZ * C_SZ, n8qk);
    const int n8v = (OUT_SZ * C_SZ) / 8;           // 32768
    cvt_f32_to_bf16_kernel<<<(n8v + 255) / 256, 256, 0, stream>>>(
        Wv, wb + (size_t)2 * DK_SZ * C_SZ, n8v);
  }

  // projection: grid (M/64, (64+64+512)/64), 4 waves/block
  dim3 g1((B_SZ * L_SZ) / 64, (2 * DK_SZ + OUT_SZ) / 64);
  qkv_proj_kernel<<<g1, dim3(128), 0, stream>>>(xb, wb, bq, bk, bv,
                                                qb, kb, vt);

  // attention: grid (L/16 query tiles, B), 4 waves/block (o-split 4x128)
  dim3 g2(L_SZ / 16, B_SZ);
  flash_attn_kernel<<<g2, dim3(128), 0, stream>>>(qb, kb, vt, out);
}
